// Windfarm_GNN_15238543966388
// MI455X (gfx1250) — matrix operand order repaired and checked by reading, not compiled, not run on hardware
//
#include <hip/hip_runtime.h>
#include <math.h>

// ---------------------------------------------------------------------------
// Windfarm GNN on MI455X (gfx1250): WMMA f16 matmuls + L2-atomic softmax agg.
// ---------------------------------------------------------------------------

typedef __attribute__((ext_vector_type(16))) _Float16 v16h;
typedef __attribute__((ext_vector_type(8)))  _Float16 v8h;
typedef __attribute__((ext_vector_type(4)))  _Float16 v4h;
typedef __attribute__((ext_vector_type(8)))  float    v8f;
typedef __attribute__((ext_vector_type(4)))  float    v4f;

#define N_NODES 50000
#define N_EDGES 1250000
#define LAT     64
#define HID     128
#define N_STEPS 5
#define EPS_GEN 1e-6f

// ---------------------------------------------------------------------------
// Weight swizzle: fp32 row-major W[K][N]  ->  f16 WMMA B fragments.
// Fragment (kt, nt) holds the 32x16 (KxN) sub-block in register order:
//   element (lane*16 + j)  <-  W[kt*32 + (lane>>4)*16 + j][nt*16 + (lane&15)]
// so a lane's v16h is 32 contiguous bytes (two b128 loads).
// ---------------------------------------------------------------------------
__global__ void swizzle_weights_kernel(const float* __restrict__ W,
                                       _Float16* __restrict__ out,
                                       int K, int Nout) {
  int idx = blockIdx.x * blockDim.x + threadIdx.x;
  int total = K * Nout;
  if (idx >= total) return;
  int frag = idx >> 9;          // 512 halfs per fragment
  int e    = idx & 511;
  int l    = e >> 4;
  int j    = e & 15;
  int ntiles = Nout >> 4;
  int kt = frag / ntiles;
  int nt = frag - kt * ntiles;
  int k  = kt * 32 + (l >> 4) * 16 + j;
  int n  = nt * 16 + (l & 15);
  out[idx] = (_Float16)W[k * Nout + n];
}

// ---------------------------------------------------------------------------
// Fragment loaders
// ---------------------------------------------------------------------------
__device__ __forceinline__ v16h ld_b_frag(const _Float16* __restrict__ Wf,
                                          int fragIdx, int lane) {
  const _Float16* p = Wf + ((size_t)fragIdx << 9) + (lane << 4);
  union { v16h v; v8h h[2]; } u;
  u.h[0] = *(const v8h*)(p);      // global_load_b128
  u.h[1] = *(const v8h*)(p + 8);  // global_load_b128
  return u.v;
}

// A (16x32, MxK) fragment from an LDS tile with row stride `stride` halfs.
// lane<16: row=lane,    K = kt*32 + {0..15}
// lane>=16: row=lane-16, K = kt*32 + {16..31}
__device__ __forceinline__ v16h ld_a_frag(const _Float16* hin, int stride,
                                          int kt, int lane) {
  const int row = lane & 15;
  const int hi  = (lane >> 4) << 4;            // 0 or 16
  const _Float16* p = hin + row * stride + (kt << 5) + hi;
  union { v16h v; v8h h[2]; } u;
  u.h[0] = *(const v8h*)(p);       // ds_load_b128
  u.h[1] = *(const v8h*)(p + 8);   // ds_load_b128
  return u.v;
}

// ---------------------------------------------------------------------------
// One dense layer of a per-wave 16-row tile:
//   out[16][NT*16] = act(hin[16][KT*32] @ W + b)
// OUTMODE: 0 = LDS f16, 1 = global f16, 2 = global f32
// ---------------------------------------------------------------------------
template<int KT, int NT, bool RELU, int OUTMODE>
__device__ __forceinline__ void mlp_layer(const _Float16* hin, int sIn,
                                          const _Float16* __restrict__ Wf,
                                          const float* __restrict__ bias,
                                          _Float16* out_h, float* out_f,
                                          int sOut, int lane) {
  v16h a[KT];
#pragma unroll
  for (int kt = 0; kt < KT; ++kt) a[kt] = ld_a_frag(hin, sIn, kt, lane);

  const int col   = lane & 15;
  const int rbase = (lane >> 4) << 3;   // 0 or 8
#pragma unroll
  for (int nt = 0; nt < NT; ++nt) {
    v8f acc = {};
#pragma unroll
    for (int kt = 0; kt < KT; ++kt) {
      v16h b = ld_b_frag(Wf, kt * NT + nt, lane);
      acc = __builtin_amdgcn_wmma_f32_16x16x32_f16(
          /*neg_a=*/false, a[kt], /*neg_b=*/false, b,
          /*c_mod=*/(short)0, acc, /*reuse_a=*/false, /*reuse_b=*/false);
    }
    float bv = bias[nt * 16 + col];
#pragma unroll
    for (int v = 0; v < 8; ++v) {
      float x = acc[v] + bv;
      if (RELU) x = fmaxf(x, 0.0f);
      int row  = rbase + v;
      int cidx = nt * 16 + col;
      if constexpr (OUTMODE == 0) out_h[row * sOut + cidx] = (_Float16)x;
      else if constexpr (OUTMODE == 1) out_h[row * sOut + cidx] = (_Float16)x;
      else out_f[row * sOut + cidx] = x;
    }
  }
}

// ---------------------------------------------------------------------------
// Encoder: IN_F -> 128 (VALU, relu) -> 128 (WMMA, relu) -> 64 (WMMA, linear)
// One wave per 16-row tile; 8 waves per block.
// ---------------------------------------------------------------------------
template<int IN_F, bool OUT32>
__global__ void __launch_bounds__(256) encoder_kernel(
    const float* __restrict__ x, int rows,
    const float* __restrict__ W0, const float* __restrict__ b0,
    const _Float16* __restrict__ W1f, const float* __restrict__ b1,
    const _Float16* __restrict__ W2f, const float* __restrict__ b2,
    _Float16* __restrict__ out_h, float* __restrict__ out_f) {
  __shared__ _Float16 sh0[8][16 * 136];
  __shared__ _Float16 sh1[8][16 * 136];
  const int lane = threadIdx.x & 31;
  const int wv   = threadIdx.x >> 5;
  const int tile = blockIdx.x * 8 + wv;
  if (tile * 16 >= rows) return;      // whole wave exits uniformly
  const int r0 = tile * 16;
  _Float16* h0 = sh0[wv];
  _Float16* h1 = sh1[wv];

  // layer 0: tiny K (2 or 3), VALU. 2 lanes per row, 64 cols each.
  {
    const int row = lane >> 1;
    const int cb  = (lane & 1) * 64;
    float xv[IN_F];
#pragma unroll
    for (int f = 0; f < IN_F; ++f) xv[f] = x[(r0 + row) * IN_F + f];
    for (int c0 = 0; c0 < 64; c0 += 8) {
      v8h pk;
#pragma unroll
      for (int j = 0; j < 8; ++j) {
        int c = cb + c0 + j;
        float acc = b0[c];
#pragma unroll
        for (int f = 0; f < IN_F; ++f) acc = fmaf(xv[f], W0[f * HID + c], acc);
        pk[j] = (_Float16)fmaxf(acc, 0.0f);
      }
      *(v8h*)&h0[row * 136 + cb + c0] = pk;
    }
  }
  __builtin_amdgcn_wave_barrier();
  mlp_layer<4, 8, true, 0>(h0, 136, W1f, b1, h1, nullptr, 136, lane);
  __builtin_amdgcn_wave_barrier();
  if constexpr (OUT32)
    mlp_layer<4, 4, false, 2>(h1, 136, W2f, b2, nullptr, out_f + (size_t)r0 * LAT, LAT, lane);
  else
    mlp_layer<4, 4, false, 1>(h1, 136, W2f, b2, out_h + (size_t)r0 * LAT, nullptr, LAT, lane);
}

// ---------------------------------------------------------------------------
// Softmax-aggregation stats (smax / denom / num), per (receiver, feature).
// ---------------------------------------------------------------------------
__global__ void __launch_bounds__(256) zero3_kernel(float* a, float* b, float* c, int n) {
  int i = blockIdx.x * blockDim.x + threadIdx.x;
  if (i < n) { a[i] = 0.0f; b[i] = 0.0f; c[i] = 0.0f; }
}

// m = relu(n[s] + e) + eps  (strictly > 0, so uint atomicMax is order-exact)
__global__ void __launch_bounds__(256) edge_max_kernel(
    const float* __restrict__ nlat, const _Float16* __restrict__ elat,
    const int* __restrict__ senders, const int* __restrict__ receivers,
    unsigned int* __restrict__ smax_u) {
  int tid = blockIdx.x * blockDim.x + threadIdx.x;  // E * 16 threads
  int e = tid >> 4;
  if (e >= N_EDGES) return;
  int f = (tid & 15) << 2;
  int s = senders[e];
  int r = receivers[e];
  v4f nv = *(const v4f*)(nlat + (size_t)s * LAT + f);
  v4h ev = *(const v4h*)(elat + (size_t)e * LAT + f);
#pragma unroll
  for (int j = 0; j < 4; ++j) {
    float m = fmaxf(nv[j] + (float)ev[j], 0.0f) + EPS_GEN;
    atomicMax(&smax_u[(size_t)r * LAT + f + j], __float_as_uint(m));
  }
}

__global__ void __launch_bounds__(256) edge_sum_kernel(
    const float* __restrict__ nlat, const _Float16* __restrict__ elat,
    const int* __restrict__ senders, const int* __restrict__ receivers,
    const float* __restrict__ smax, float* __restrict__ den,
    float* __restrict__ num) {
  int tid = blockIdx.x * blockDim.x + threadIdx.x;
  int e = tid >> 4;
  if (e >= N_EDGES) return;
  int f = (tid & 15) << 2;
  int s = senders[e];
  int r = receivers[e];
  v4f nv = *(const v4f*)(nlat + (size_t)s * LAT + f);
  v4h ev = *(const v4h*)(elat + (size_t)e * LAT + f);
  v4f sv = *(const v4f*)(smax + (size_t)r * LAT + f);
#pragma unroll
  for (int j = 0; j < 4; ++j) {
    float m  = fmaxf(nv[j] + (float)ev[j], 0.0f) + EPS_GEN;
    float ex = __expf(m - sv[j]);
    atomicAdd(&den[(size_t)r * LAT + f + j], ex);
    atomicAdd(&num[(size_t)r * LAT + f + j], ex * m);
  }
}

// ---------------------------------------------------------------------------
// Node update: x = n + num/den, then 64->128->128->64 MLP (WMMA).
// ---------------------------------------------------------------------------
__global__ void __launch_bounds__(256) node_update_kernel(
    const float* __restrict__ nin, const float* __restrict__ num,
    const float* __restrict__ den,
    const _Float16* __restrict__ W0f, const float* __restrict__ b0,
    const _Float16* __restrict__ W1f, const float* __restrict__ b1,
    const _Float16* __restrict__ W2f, const float* __restrict__ b2,
    float* __restrict__ nout) {
  __shared__ _Float16 sx[8][16 * 72];
  __shared__ _Float16 sh0[8][16 * 136];
  __shared__ _Float16 sh1[8][16 * 136];
  const int lane = threadIdx.x & 31;
  const int wv   = threadIdx.x >> 5;
  const int tile = blockIdx.x * 8 + wv;
  if (tile * 16 >= N_NODES) return;
  const int r0 = tile * 16;
  _Float16* hx = sx[wv];
  _Float16* h0 = sh0[wv];
  _Float16* h1 = sh1[wv];

  // x = n + (den>0 ? num/den : 0), f16 into LDS. 2 lanes/row, 32 cols each.
  {
    const int row = lane >> 1;
    const int cb  = (lane & 1) * 32;
    const int gb  = (r0 + row) * LAT + cb;
    for (int c0 = 0; c0 < 32; c0 += 8) {
      v8h pk;
#pragma unroll
      for (int j = 0; j < 8; ++j) {
        int gi = gb + c0 + j;
        float d   = den[gi];
        float agg = (d > 0.0f) ? (num[gi] / d) : 0.0f;
        pk[j] = (_Float16)(nin[gi] + agg);
      }
      *(v8h*)&hx[row * 72 + cb + c0] = pk;
    }
  }
  __builtin_amdgcn_wave_barrier();
  mlp_layer<2, 8, true, 0>(hx, 72, W0f, b0, h0, nullptr, 136, lane);
  __builtin_amdgcn_wave_barrier();
  mlp_layer<4, 8, true, 0>(h0, 136, W1f, b1, h1, nullptr, 136, lane);
  __builtin_amdgcn_wave_barrier();
  mlp_layer<4, 4, false, 2>(h1, 136, W2f, b2, nullptr, nout + (size_t)r0 * LAT, LAT, lane);
}

// ---------------------------------------------------------------------------
// Decoder: 64->128->128 (WMMA) -> 2 (VALU) + zero-input mask.
// ---------------------------------------------------------------------------
__global__ void __launch_bounds__(256) decoder_kernel(
    const float* __restrict__ nin, const float* __restrict__ nodes_raw,
    const _Float16* __restrict__ W0f, const float* __restrict__ b0,
    const _Float16* __restrict__ W1f, const float* __restrict__ b1,
    const float* __restrict__ W2, const float* __restrict__ b2,
    float* __restrict__ out) {
  __shared__ _Float16 sx[8][16 * 72];
  __shared__ _Float16 sh0[8][16 * 136];
  __shared__ _Float16 sh1[8][16 * 136];
  const int lane = threadIdx.x & 31;
  const int wv   = threadIdx.x >> 5;
  const int tile = blockIdx.x * 8 + wv;
  if (tile * 16 >= N_NODES) return;
  const int r0 = tile * 16;
  _Float16* hx = sx[wv];
  _Float16* h0 = sh0[wv];
  _Float16* h1 = sh1[wv];

  {
    const int row = lane >> 1;
    const int cb  = (lane & 1) * 32;
    const int gb  = (r0 + row) * LAT + cb;
    for (int c0 = 0; c0 < 32; c0 += 8) {
      v8h pk;
#pragma unroll
      for (int j = 0; j < 8; ++j) pk[j] = (_Float16)nin[gb + c0 + j];
      *(v8h*)&hx[row * 72 + cb + c0] = pk;
    }
  }
  __builtin_amdgcn_wave_barrier();
  mlp_layer<2, 8, true, 0>(hx, 72, W0f, b0, h0, nullptr, 136, lane);
  __builtin_amdgcn_wave_barrier();
  mlp_layer<4, 8, true, 0>(h0, 136, W1f, b1, h1, nullptr, 136, lane);
  __builtin_amdgcn_wave_barrier();
  // final 128 -> 2, one (row, target) pair per lane
  {
    const int row = lane >> 1;
    const int t   = lane & 1;
    float acc = b2[t];
    for (int k0 = 0; k0 < HID; k0 += 8) {
      v8h hk = *(const v8h*)&h1[row * 136 + k0];
#pragma unroll
      for (int j = 0; j < 8; ++j)
        acc = fmaf((float)hk[j], W2[(k0 + j) * 2 + t], acc);
    }
    int g = r0 + row;
    float m = fabsf(nodes_raw[g * 2]) + fabsf(nodes_raw[g * 2 + 1]);
    out[g * 2 + t] = (m != 0.0f) ? acc : 0.0f;
  }
}

// ---------------------------------------------------------------------------
// Host-side orchestration
// ---------------------------------------------------------------------------
extern "C" void kernel_launch(void* const* d_in, const int* in_sizes, int n_in,
                              void* d_out, int out_size, void* d_ws, size_t ws_size,
                              hipStream_t stream) {
  (void)in_sizes; (void)n_in; (void)out_size; (void)ws_size;

  const float* nodes     = (const float*)d_in[0];
  const float* edges     = (const float*)d_in[1];
  const int*   senders   = (const int*)d_in[2];
  const int*   receivers = (const int*)d_in[3];
  const float* enW0 = (const float*)d_in[4];  const float* enb0 = (const float*)d_in[5];
  const float* enW1 = (const float*)d_in[6];  const float* enb1 = (const float*)d_in[7];
  const float* enW2 = (const float*)d_in[8];  const float* enb2 = (const float*)d_in[9];
  const float* eeW0 = (const float*)d_in[10]; const float* eeb0 = (const float*)d_in[11];
  const float* eeW1 = (const float*)d_in[12]; const float* eeb1 = (const float*)d_in[13];
  const float* eeW2 = (const float*)d_in[14]; const float* eeb2 = (const float*)d_in[15];
  const float* pW0  = (const float*)d_in[16]; const float* pb0  = (const float*)d_in[17];
  const float* pW1  = (const float*)d_in[18]; const float* pb1  = (const float*)d_in[19];
  const float* pW2  = (const float*)d_in[20]; const float* pb2  = (const float*)d_in[21];
  const float* dW0  = (const float*)d_in[22]; const float* db0  = (const float*)d_in[23];
  const float* dW1  = (const float*)d_in[24]; const float* db1  = (const float*)d_in[25];
  const float* dW2  = (const float*)d_in[26]; const float* db2  = (const float*)d_in[27];
  float* out = (float*)d_out;

  // --- workspace carve-up ---
  size_t off = 0;
  char* base = (char*)d_ws;
  auto carve = [&](size_t bytes) -> char* {
    char* p = base + off;
    off += (bytes + 255) & ~(size_t)255;
    return p;
  };
  _Float16* e_lat = (_Float16*)carve((size_t)N_EDGES * LAT * sizeof(_Float16));
  float*    n_a   = (float*)carve((size_t)N_NODES * LAT * sizeof(float));
  float*    n_b   = (float*)carve((size_t)N_NODES * LAT * sizeof(float));
  float*    smax  = (float*)carve((size_t)N_NODES * LAT * sizeof(float));
  float*    den   = (float*)carve((size_t)N_NODES * LAT * sizeof(float));
  float*    num   = (float*)carve((size_t)N_NODES * LAT * sizeof(float));
  _Float16* enW1f = (_Float16*)carve((size_t)HID * HID * sizeof(_Float16));
  _Float16* enW2f = (_Float16*)carve((size_t)HID * LAT * sizeof(_Float16));
  _Float16* eeW1f = (_Float16*)carve((size_t)HID * HID * sizeof(_Float16));
  _Float16* eeW2f = (_Float16*)carve((size_t)HID * LAT * sizeof(_Float16));
  _Float16* pW0f  = (_Float16*)carve((size_t)N_STEPS * LAT * HID * sizeof(_Float16));
  _Float16* pW1f  = (_Float16*)carve((size_t)N_STEPS * HID * HID * sizeof(_Float16));
  _Float16* pW2f  = (_Float16*)carve((size_t)N_STEPS * HID * LAT * sizeof(_Float16));
  _Float16* dW0f  = (_Float16*)carve((size_t)LAT * HID * sizeof(_Float16));
  _Float16* dW1f  = (_Float16*)carve((size_t)HID * HID * sizeof(_Float16));

  auto swz = [&](const float* W, _Float16* dst, int K, int Nout) {
    int total = K * Nout;
    swizzle_weights_kernel<<<(total + 255) / 256, 256, 0, stream>>>(W, dst, K, Nout);
  };
  swz(enW1, enW1f, HID, HID);
  swz(enW2, enW2f, HID, LAT);
  swz(eeW1, eeW1f, HID, HID);
  swz(eeW2, eeW2f, HID, LAT);
  for (int s = 0; s < N_STEPS; ++s) {
    swz(pW0 + (size_t)s * LAT * HID, pW0f + (size_t)s * LAT * HID, LAT, HID);
    swz(pW1 + (size_t)s * HID * HID, pW1f + (size_t)s * HID * HID, HID, HID);
    swz(pW2 + (size_t)s * HID * LAT, pW2f + (size_t)s * HID * LAT, HID, LAT);
  }
  swz(dW0, dW0f, LAT, HID);
  swz(dW1, dW1f, HID, HID);

  // --- encoders ---
  const int nodeTiles = N_NODES / 16;           // 3125
  const int edgeTiles = N_EDGES / 16;           // 78125
  const int nodeBlocks = (nodeTiles + 7) / 8;   // 391
  const int edgeBlocks = (edgeTiles + 7) / 8;   // 9766
  encoder_kernel<2, true><<<nodeBlocks, 256, 0, stream>>>(
      nodes, N_NODES, enW0, enb0, enW1f, enb1, enW2f, enb2, nullptr, n_a);
  encoder_kernel<3, false><<<edgeBlocks, 256, 0, stream>>>(
      edges, N_EDGES, eeW0, eeb0, eeW1f, eeb1, eeW2f, eeb2, e_lat, nullptr);

  // --- processor steps ---
  const int statN = N_NODES * LAT;
  const int edgeThreads = N_EDGES * 16;
  float* cur = n_a;
  float* nxt = n_b;
  for (int s = 0; s < N_STEPS; ++s) {
    zero3_kernel<<<(statN + 255) / 256, 256, 0, stream>>>(smax, den, num, statN);
    edge_max_kernel<<<(edgeThreads + 255) / 256, 256, 0, stream>>>(
        cur, e_lat, senders, receivers, (unsigned int*)smax);
    edge_sum_kernel<<<(edgeThreads + 255) / 256, 256, 0, stream>>>(
        cur, e_lat, senders, receivers, smax, den, num);
    node_update_kernel<<<nodeBlocks, 256, 0, stream>>>(
        cur, num, den,
        pW0f + (size_t)s * LAT * HID, pb0 + (size_t)s * HID,
        pW1f + (size_t)s * HID * HID, pb1 + (size_t)s * HID,
        pW2f + (size_t)s * HID * LAT, pb2 + (size_t)s * LAT,
        nxt);
    float* tmp = cur; cur = nxt; nxt = tmp;
  }

  // --- decoder ---
  decoder_kernel<<<nodeBlocks, 256, 0, stream>>>(
      cur, nodes, dW0f, db0, dW1f, db1, dW2, db2, out);
}
